// DIFMultiHeadAttention_26826365731124
// MI455X (gfx1250) — matrix operand order repaired and checked by reading, not compiled, not run on hardware
//
#include <hip/hip_runtime.h>
#include <hip/hip_bf16.h>

// ---------------------------------------------------------------------------
// DIF multi-head attention for MI455X (gfx1250, wave32, WMMA).
// All GEMMs run on v_wmma_f32_16x16x32_f16 with f32 accumulation.
// Weights are pre-converted to f16 once; inner GEMM loops are cvt-free.
// ---------------------------------------------------------------------------

typedef __attribute__((ext_vector_type(16))) _Float16 v16h;
typedef __attribute__((ext_vector_type(8)))  _Float16 v8h;
typedef __attribute__((ext_vector_type(4)))  _Float16 v4h;
typedef __attribute__((ext_vector_type(8)))  float    v8f;
typedef __attribute__((ext_vector_type(4)))  float    v4f;

#define BB   64
#define SS   256
#define HDIM 512
#define NHEAD 8
#define DHEAD 64
#define AHDIM 256
#define DATTR 32
#define MM   (BB * SS)          // 16384 rows

__device__ __forceinline__ v8f wmma_f16(v16h a, v16h b, v8f c) {
  // D = A x B + C, 16x16x32, f16 in / f32 acc
  return __builtin_amdgcn_wmma_f32_16x16x32_f16(
      /*neg_a=*/false, a, /*neg_b=*/false, b,
      /*c_mod=*/(short)0, c, /*reuse_a=*/false, /*reuse_b=*/false);
}

// A-fragment (16x32 f16): lane holds row M=lane%16.
// halves 0..7  = K[(lane>=16?16:0) + 0..7]
// halves 8..15 = K[(lane>=16?16:0) + 8..15]
__device__ __forceinline__ v16h load_a(const _Float16* row, int lane) {
  const int c0 = (lane & 16) ? 16 : 0;
  v8h lo = *(const v8h*)(row + c0);
  v8h hi = *(const v8h*)(row + c0 + 8);
  return __builtin_shufflevector(lo, hi, 0,1,2,3,4,5,6,7,8,9,10,11,12,13,14,15);
}

// ---------------------------------------------------------------------------
// K0: f32 -> f16 conversion (weights, fw1)
// ---------------------------------------------------------------------------
__global__ void cvt_f16_kernel(const float* __restrict__ src,
                               _Float16* __restrict__ dst, int n) {
  int i = blockIdx.x * blockDim.x + threadIdx.x;
  if (i < n) dst[i] = (_Float16)src[i];
}

// ---------------------------------------------------------------------------
// K1: Y[m,n](f16) = X[m,:]·W[n,:] + bias[n]   (X:[M,K] f32, W:[N,K] f16)
// grid: (M/16, N/256), block 256 (8 waves, TWO 16-wide n-tiles per wave).
// K-step 64: two A fragments per staging barrier -> 4 wmma per barrier pair.
// B-fragments: single aligned 32B v16h loads (no cvt in the hot loop).
// Optional VT: also store output transposed as [b,h,d,s] (used for V).
// ---------------------------------------------------------------------------
__global__ void proj_kernel(const float* __restrict__ X,
                            const _Float16* __restrict__ W,
                            const float* __restrict__ bias,
                            _Float16* __restrict__ Y,
                            _Float16* __restrict__ VT,   // nullable
                            int N, int K) {
  __shared__ _Float16 Xs[16][64];      // 2 KB
  const int m0   = blockIdx.x * 16;
  const int tid  = threadIdx.x;
  const int w    = tid >> 5, lane = tid & 31;
  const int lm   = lane & 15, hi16 = lane >> 4;
  const int offB = hi16 ? 16 : 0;
  const int n0   = blockIdx.y * 256 + w * 32;   // tiles n0 and n0+16

  const int xrow = tid >> 4;           // 0..15
  const int xcol = (tid & 15) * 4;     // 0..60

  v8f c0 = {}, c1 = {};
  const int ksteps = K >> 6;
  for (int ks = 0; ks < ksteps; ++ks) {
    {  // stage X[16][64] chunk as f16 (vector load + packed cvt + b64 store)
      v4f xv = *(const v4f*)(X + (size_t)(m0 + xrow) * K + ks * 64 + xcol);
      v4h hx;
#pragma unroll
      for (int i = 0; i < 4; ++i) hx[i] = (_Float16)xv[i];
      *(v4h*)&Xs[xrow][xcol] = hx;
    }
    __syncthreads();
    const _Float16* wr0 = W + (size_t)(n0 + lm) * K + ks * 64 + offB;
    const _Float16* wr1 = W + (size_t)(n0 + 16 + lm) * K + ks * 64 + offB;
    __builtin_prefetch(wr0 + 64, 0, 1);   // stream next W chunk toward L2
    v16h a0 = load_a(&Xs[lm][0], lane);
    v16h a1 = load_a(&Xs[lm][32], lane);
    c0 = wmma_f16(a0, *(const v16h*)wr0,        c0);
    c0 = wmma_f16(a1, *(const v16h*)(wr0 + 32), c0);
    c1 = wmma_f16(a0, *(const v16h*)wr1,        c1);
    c1 = wmma_f16(a1, *(const v16h*)(wr1 + 32), c1);
    __syncthreads();
  }

  const int col0 = n0 + lm, col1 = n0 + 16 + lm;
  const float bv0 = bias[col0], bv1 = bias[col1];
#pragma unroll
  for (int r = 0; r < 8; ++r) {
    const int row = r + hi16 * 8;
    const int m = m0 + row;
    const _Float16 y0 = (_Float16)(c0[r] + bv0);
    const _Float16 y1 = (_Float16)(c1[r] + bv1);
    Y[(size_t)m * N + col0] = y0;
    Y[(size_t)m * N + col1] = y1;
    if (VT) {  // transposed copy: VT[((b*8+h)*64 + d)*256 + s]
      const int b = m >> 8, s = m & 255;
      VT[(((size_t)(b * 8 + (col0 >> 6))) * 64 + (col0 & 63)) * SS + s] = y0;
      VT[(((size_t)(b * 8 + (col1 >> 6))) * 64 + (col1 & 63)) * SS + s] = y1;
    }
  }
}

// ---------------------------------------------------------------------------
// K2: stacked scores SC[b,h,src,q,n] (f16), src: 0=attr0,1=attr1,2=item,3=pos
// grid: (B*H, S/16), block 256: wave = (src, n-tile half)
// ---------------------------------------------------------------------------
__global__ void scores_kernel(const _Float16* __restrict__ Qh,
                              const _Float16* __restrict__ Kh,
                              const _Float16* __restrict__ QPh,
                              const _Float16* __restrict__ KPh,
                              const _Float16* __restrict__ QA,
                              const _Float16* __restrict__ KA,
                              _Float16* __restrict__ SC) {
  const int bh = blockIdx.x, b = bh >> 3, h = bh & 7;
  const int q0 = blockIdx.y * 16;
  const int tid = threadIdx.x;
  const int w = tid >> 5, lane = tid & 31;
  const int lm = lane & 15, hi16 = lane >> 4;
  const int offB = hi16 ? 16 : 0;
  const int src = w & 3, part = w >> 2;

  const size_t qrow = (size_t)(b * SS + q0 + lm);
  v16h a0 = {}, a1 = {};
  if (src == 2) {
    const _Float16* qr = Qh + qrow * HDIM + h * DHEAD;
    a0 = load_a(qr, lane); a1 = load_a(qr + 32, lane);
  } else if (src == 3) {
    const _Float16* qr = QPh + qrow * HDIM + h * DHEAD;
    a0 = load_a(qr, lane); a1 = load_a(qr + 32, lane);
  } else {
    const _Float16* qr = QA + ((size_t)src * MM + qrow) * AHDIM + h * DATTR;
    a0 = load_a(qr, lane);
  }

  for (int tt = 0; tt < 8; ++tt) {
    const int n0 = (part * 8 + tt) * 16;
    const size_t krow = (size_t)(b * SS + n0 + lm);
    v8f c = {};
    if (src >= 2) {
      const _Float16* kr = (src == 2 ? Kh : KPh) + krow * HDIM + h * DHEAD + offB;
      c = wmma_f16(a0, *(const v16h*)kr, c);
      c = wmma_f16(a1, *(const v16h*)(kr + 32), c);
    } else {
      const _Float16* kr = KA + ((size_t)src * MM + krow) * AHDIM + h * DATTR + offB;
      c = wmma_f16(a0, *(const v16h*)kr, c);
    }
    _Float16* out = SC + (((size_t)bh * 4 + src) * SS + q0) * SS + n0;
#pragma unroll
    for (int r = 0; r < 8; ++r)
      out[(size_t)(r + hi16 * 8) * SS + lm] = (_Float16)c[r];
  }
}

// ---------------------------------------------------------------------------
// K3: gate energy GEMM + gate softmax + fused scores + row softmax + ctx
// grid: (B*H, S/16), block 256 (8 waves)
// ---------------------------------------------------------------------------
__global__ void gate_ctx_kernel(const _Float16* __restrict__ SC,
                                const _Float16* __restrict__ VT,
                                const _Float16* __restrict__ fw1h,
                                const float* __restrict__ fb1,
                                const float* __restrict__ fw2,
                                _Float16* __restrict__ CTX) {
  __shared__ _Float16 scbuf[4][16][SS];   // 32 KB
  __shared__ float    sbuf[16][SS];       // 16 KB
  __shared__ _Float16 pbuf[16][SS];       //  8 KB
  __shared__ float    e_lds[4][16];
  __shared__ float    gw[16][4];
  __shared__ float    red[16][16];
  __shared__ float    rowmax[16];
  __shared__ float    rowsum[16];

  const int bh = blockIdx.x, b = bh >> 3, h = bh & 7;
  const int q0 = blockIdx.y * 16;
  const int tid = threadIdx.x;
  const int w = tid >> 5, lane = tid & 31;
  const int lm = lane & 15, hi16 = lane >> 4;
  const int offB = hi16 ? 16 : 0;

  // Stage all 4 score strips [4][16][256] into LDS (16-byte chunks).
  const size_t scbase = (size_t)bh * 4 * (SS * SS) + (size_t)q0 * SS;
#pragma unroll
  for (int it = 0; it < 8; ++it) {
    int cch = tid + it * 256;                // 0..2047
    int j = cch >> 9, rem = cch & 511;
    int row = rem >> 5, cc = (rem & 31) * 8;
    *(v8h*)&scbuf[j][row][cc] =
        *(const v8h*)(SC + scbase + (size_t)j * (SS * SS) + (size_t)row * SS + cc);
  }
  if (tid < 64) e_lds[tid >> 4][tid & 15] = 0.f;
  __syncthreads();

  // ---- gate energy: E_j = relu(SC_j @ fw1^T + fb1) · fw2 ----
  const int src = w & 3, part = w >> 2;
  float ep[8];
#pragma unroll
  for (int r = 0; r < 8; ++r) ep[r] = 0.f;

  for (int tt = 0; tt < 8; ++tt) {
    const int n0 = (part + tt * 2) * 16;
    v8f c = {};
#pragma unroll
    for (int kc = 0; kc < 8; ++kc) {
      v16h a  = load_a(&scbuf[src][lm][kc * 32], lane);
      v16h bb = *(const v16h*)(fw1h + (size_t)(n0 + lm) * SS + kc * 32 + offB);
      c = wmma_f16(a, bb, c);
    }
    const int col = n0 + lm;
    const float w2 = fw2[col], b1 = fb1[col];
#pragma unroll
    for (int r = 0; r < 8; ++r) {
      float v = c[r] + b1;
      v = v > 0.f ? v : 0.f;
      ep[r] += v * w2;
    }
  }
#pragma unroll
  for (int r = 0; r < 8; ++r) {          // reduce over the 16 columns held per half
    float v = ep[r];
    for (int m = 1; m < 16; m <<= 1) v += __shfl_xor(v, m, 32);
    if (lm == 0) atomicAdd(&e_lds[src][r + hi16 * 8], v);
  }
  __syncthreads();

  // ---- gate softmax over 4 sources (fb2 cancels) ----
  if (tid < 16) {
    float m = e_lds[0][tid];
    for (int j = 1; j < 4; ++j) m = fmaxf(m, e_lds[j][tid]);
    float s = 0.f, e[4];
    for (int j = 0; j < 4; ++j) { e[j] = __expf(e_lds[j][tid] - m); s += e[j]; }
    float inv = 1.f / s;
    for (int j = 0; j < 4; ++j) gw[tid][j] = e[j] * inv;
  }
  __syncthreads();

  // ---- fused scores = sum_j gw_j * SC_j, scaled by 1/sqrt(DH) ----
  {
    const int row = tid >> 4, c0 = (tid & 15) * 16;
    const float g0 = gw[row][0], g1 = gw[row][1], g2 = gw[row][2], g3 = gw[row][3];
    const float invsq = 0.125f;
    for (int k = 0; k < 16; ++k) {
      int cc = c0 + k;
      float v = g0 * (float)scbuf[0][row][cc] + g1 * (float)scbuf[1][row][cc] +
                g2 * (float)scbuf[2][row][cc] + g3 * (float)scbuf[3][row][cc];
      sbuf[row][cc] = v * invsq;
    }
  }
  __syncthreads();

  // ---- row softmax (unnormalized probs in f16; 1/rowsum folded into ctx) ----
  {
    const int row = tid >> 4, t = tid & 15, c0 = t * 16;
    float m = -1e30f;
    for (int k = 0; k < 16; ++k) m = fmaxf(m, sbuf[row][c0 + k]);
    red[row][t] = m;
  }
  __syncthreads();
  if (tid < 16) {
    float m = red[tid][0];
    for (int t = 1; t < 16; ++t) m = fmaxf(m, red[tid][t]);
    rowmax[tid] = m;
  }
  __syncthreads();
  {
    const int row = tid >> 4, t = tid & 15, c0 = t * 16;
    const float m = rowmax[row];
    float s = 0.f;
    for (int k = 0; k < 16; ++k) {
      float e = __expf(sbuf[row][c0 + k] - m);
      pbuf[row][c0 + k] = (_Float16)e;
      s += e;
    }
    red[row][t] = s;
  }
  __syncthreads();
  if (tid < 16) {
    float s = 0.f;
    for (int t = 0; t < 16; ++t) s += red[tid][t];
    rowsum[tid] = s;
  }
  __syncthreads();

  // ---- ctx = probs @ V : waves 0..3, one 16-wide d-tile each.
  //      B-fragments are contiguous v16h loads from the transposed V copy.
  if (w < 4) {
    v8f c = {};
    const _Float16* vrow = VT + ((size_t)bh * DHEAD + w * 16 + lm) * SS + offB;
#pragma unroll
    for (int kc = 0; kc < 8; ++kc) {
      v16h a  = load_a(&pbuf[lm][kc * 32], lane);
      v16h bb = *(const v16h*)(vrow + kc * 32);
      c = wmma_f16(a, bb, c);
    }
    const int col = h * DHEAD + w * 16 + lm;
#pragma unroll
    for (int r = 0; r < 8; ++r) {
      int row = r + hi16 * 8;
      CTX[(size_t)(b * SS + q0 + row) * HDIM + col] = (_Float16)(c[r] / rowsum[row]);
    }
  }
}

// ---------------------------------------------------------------------------
// K4: out = LN(ctx @ Wd^T + bd + input) ; grid: M/16, block 256 (Wd in f16)
// ---------------------------------------------------------------------------
__global__ void outproj_ln_kernel(const _Float16* __restrict__ CTX,
                                  const _Float16* __restrict__ Wd,
                                  const float* __restrict__ bd,
                                  const float* __restrict__ input,
                                  const float* __restrict__ gamma,
                                  const float* __restrict__ beta,
                                  float* __restrict__ out) {
  __shared__ _Float16 ctxs[16][HDIM];  // 16 KB
  __shared__ float    hbuf[16][HDIM];  // 32 KB
  __shared__ float    red[2][16][16];
  __shared__ float    stats[16][2];

  const int m0 = blockIdx.x * 16;
  const int tid = threadIdx.x;
  const int w = tid >> 5, lane = tid & 31;
  const int lm = lane & 15, hi16 = lane >> 4;
  const int offB = hi16 ? 16 : 0;

#pragma unroll
  for (int it = 0; it < 4; ++it) {
    int cch = tid + it * 256;                 // 0..1023 chunks of 8
    int row = cch >> 6, cc = (cch & 63) * 8;
    *(v8h*)&ctxs[row][cc] = *(const v8h*)(CTX + (size_t)(m0 + row) * HDIM + cc);
  }
  __syncthreads();

  for (int tt = 0; tt < 4; ++tt) {
    const int n0 = (w + tt * 8) * 16;
    v8f c = {};
#pragma unroll
    for (int kc = 0; kc < 16; ++kc) {
      v16h a  = load_a(&ctxs[lm][kc * 32], lane);
      v16h bb = *(const v16h*)(Wd + (size_t)(n0 + lm) * HDIM + kc * 32 + offB);
      c = wmma_f16(a, bb, c);
    }
    const int col = n0 + lm;
    const float bias = bd[col];
#pragma unroll
    for (int r = 0; r < 8; ++r) {
      int row = r + hi16 * 8;
      hbuf[row][col] = c[r] + bias + input[(size_t)(m0 + row) * HDIM + col];
    }
  }
  __syncthreads();
  {
    const int row = tid >> 4, t = tid & 15, c0 = t * 32;
    float s = 0.f, ss = 0.f;
    for (int k = 0; k < 32; ++k) { float v = hbuf[row][c0 + k]; s += v; ss += v * v; }
    red[0][row][t] = s; red[1][row][t] = ss;
  }
  __syncthreads();
  if (tid < 16) {
    float s = 0.f, ss = 0.f;
    for (int t = 0; t < 16; ++t) { s += red[0][tid][t]; ss += red[1][tid][t]; }
    float mu = s * (1.f / 512.f);
    float var = ss * (1.f / 512.f) - mu * mu;
    stats[tid][0] = mu; stats[tid][1] = rsqrtf(var + 1e-12f);
  }
  __syncthreads();
  {
    const int row = tid >> 4, t = tid & 15, c0 = t * 32;
    const float mu = stats[row][0], rs = stats[row][1];
    for (int k = 0; k < 32; ++k) {
      int cc = c0 + k;
      out[(size_t)(m0 + row) * HDIM + cc] =
          gamma[cc] * (hbuf[row][cc] - mu) * rs + beta[cc];
    }
  }
}

// ---------------------------------------------------------------------------
extern "C" void kernel_launch(void* const* d_in, const int* in_sizes, int n_in,
                              void* d_out, int out_size, void* d_ws, size_t ws_size,
                              hipStream_t stream) {
  (void)in_sizes; (void)n_in; (void)out_size; (void)ws_size;

  const float* input = (const float*)d_in[0];
  const float* attr  = (const float*)d_in[1];
  const float* pos   = (const float*)d_in[2];
  const float* Wq  = (const float*)d_in[3],  *bq  = (const float*)d_in[4];
  const float* Wk  = (const float*)d_in[5],  *bk  = (const float*)d_in[6];
  const float* Wv  = (const float*)d_in[7],  *bv  = (const float*)d_in[8];
  const float* Wqp = (const float*)d_in[9],  *bqp = (const float*)d_in[10];
  const float* Wkp = (const float*)d_in[11], *bkp = (const float*)d_in[12];
  const float* Wqa = (const float*)d_in[13], *bqa = (const float*)d_in[14];
  const float* Wka = (const float*)d_in[15], *bka = (const float*)d_in[16];
  const float* fw1 = (const float*)d_in[17], *fb1 = (const float*)d_in[18];
  const float* fw2 = (const float*)d_in[19];
  const float* Wd  = (const float*)d_in[21], *bd  = (const float*)d_in[22];
  const float* gamma = (const float*)d_in[23], *beta = (const float*)d_in[24];
  float* out = (float*)d_out;

  // workspace carve-up (f16 elements)
  _Float16* ws16 = (_Float16*)d_ws;
  size_t off = 0;
  _Float16* Qh  = ws16 + off; off += (size_t)MM * HDIM;
  _Float16* Kh  = ws16 + off; off += (size_t)MM * HDIM;
  _Float16* Vh  = ws16 + off; off += (size_t)MM * HDIM;
  _Float16* VT  = ws16 + off; off += (size_t)MM * HDIM;   // V transposed [b,h,d,s]
  _Float16* QPh = ws16 + off; off += (size_t)MM * HDIM;
  _Float16* KPh = ws16 + off; off += (size_t)MM * HDIM;
  _Float16* QA  = ws16 + off; off += (size_t)2 * MM * AHDIM;
  _Float16* KA  = ws16 + off; off += (size_t)2 * MM * AHDIM;
  _Float16* fw1h = ws16 + off; off += (size_t)SS * SS;
  // f16 weight copies (converted once, reused by 1024 blocks each)
  _Float16* Wqh  = ws16 + off; off += (size_t)HDIM * HDIM;
  _Float16* Wkh  = ws16 + off; off += (size_t)HDIM * HDIM;
  _Float16* Wvh  = ws16 + off; off += (size_t)HDIM * HDIM;
  _Float16* Wqph = ws16 + off; off += (size_t)HDIM * HDIM;
  _Float16* Wkph = ws16 + off; off += (size_t)HDIM * HDIM;
  _Float16* Wqah = ws16 + off; off += (size_t)2 * AHDIM * AHDIM;
  _Float16* Wkah = ws16 + off; off += (size_t)2 * AHDIM * AHDIM;
  _Float16* Wdh  = ws16 + off; off += (size_t)HDIM * HDIM;
  _Float16* SC  = ws16 + off; off += (size_t)BB * NHEAD * 4 * SS * SS;
  _Float16* CTX = ws16 + off; off += (size_t)MM * HDIM;

  const dim3 blk(256);

  // ---- one-time f32 -> f16 weight conversion ----
  cvt_f16_kernel<<<dim3(256),  blk, 0, stream>>>(fw1, fw1h, SS * SS);
  cvt_f16_kernel<<<dim3(1024), blk, 0, stream>>>(Wq,  Wqh,  HDIM * HDIM);
  cvt_f16_kernel<<<dim3(1024), blk, 0, stream>>>(Wk,  Wkh,  HDIM * HDIM);
  cvt_f16_kernel<<<dim3(1024), blk, 0, stream>>>(Wv,  Wvh,  HDIM * HDIM);
  cvt_f16_kernel<<<dim3(1024), blk, 0, stream>>>(Wqp, Wqph, HDIM * HDIM);
  cvt_f16_kernel<<<dim3(1024), blk, 0, stream>>>(Wkp, Wkph, HDIM * HDIM);
  cvt_f16_kernel<<<dim3(512),  blk, 0, stream>>>(Wqa, Wqah, 2 * AHDIM * AHDIM);
  cvt_f16_kernel<<<dim3(512),  blk, 0, stream>>>(Wka, Wkah, 2 * AHDIM * AHDIM);
  cvt_f16_kernel<<<dim3(1024), blk, 0, stream>>>(Wd,  Wdh,  HDIM * HDIM);

  const dim3 gProj(MM / 16, HDIM / 256);
  proj_kernel<<<gProj, blk, 0, stream>>>(input, Wqh,  bq,  Qh,  nullptr, HDIM, HDIM);
  proj_kernel<<<gProj, blk, 0, stream>>>(input, Wkh,  bk,  Kh,  nullptr, HDIM, HDIM);
  proj_kernel<<<gProj, blk, 0, stream>>>(input, Wvh,  bv,  Vh,  VT,      HDIM, HDIM);
  proj_kernel<<<gProj, blk, 0, stream>>>(pos,   Wqph, bqp, QPh, nullptr, HDIM, HDIM);
  proj_kernel<<<gProj, blk, 0, stream>>>(pos,   Wkph, bkp, KPh, nullptr, HDIM, HDIM);

  const dim3 gAttr(MM / 16, AHDIM / 256);
  for (int f = 0; f < 2; ++f) {
    proj_kernel<<<gAttr, blk, 0, stream>>>(attr, Wqah + (size_t)f * AHDIM * AHDIM,
                                           bqa + (size_t)f * AHDIM,
                                           QA + (size_t)f * MM * AHDIM, nullptr,
                                           AHDIM, AHDIM);
    proj_kernel<<<gAttr, blk, 0, stream>>>(attr, Wkah + (size_t)f * AHDIM * AHDIM,
                                           bka + (size_t)f * AHDIM,
                                           KA + (size_t)f * MM * AHDIM, nullptr,
                                           AHDIM, AHDIM);
  }

  scores_kernel<<<dim3(BB * NHEAD, SS / 16), blk, 0, stream>>>(
      Qh, Kh, QPh, KPh, QA, KA, SC);

  gate_ctx_kernel<<<dim3(BB * NHEAD, SS / 16), blk, 0, stream>>>(
      SC, VT, fw1h, fb1, fw2, CTX);

  outproj_ln_kernel<<<dim3(MM / 16), blk, 0, stream>>>(
      CTX, Wdh, bd, input, gamma, beta, out);
}
